// SampleRNN_4861902979254
// MI455X (gfx1250) — compile-verified, matrix operand order checked by reading, not run on hardware
//
#include <hip/hip_runtime.h>

// ---------------------------------------------------------------------------
// SampleRNN 3-tier LSTM stack for MI455X (gfx1250), wave32 + WMMA bf16.
//
//  * All matrix math uses v_wmma_f32_16x16x32_bf16 (f32 accumulate).
//  * Activations/weights live in a pre-swizzled "fragment-native" layout so
//    A operand loads are single contiguous 32B/lane global reads (L2-hot).
//  * Recurrence: 16 persistent workgroups x 512 threads. Each WG stages its
//    256 KB weight slice in LDS once (CDNA5 WGP has 320 KB LDS) and holds its
//    cell state in registers for all T steps; per step it computes
//    [x_t ; h_{t-1}](32x1024) @ Wslice(1024x128) with 32 WMMAs/wave, B-operands
//    from LDS, then an agent-scope atomic barrier + acquire fence.
// ---------------------------------------------------------------------------

typedef __attribute__((ext_vector_type(16))) __bf16 v16bf;
typedef __attribute__((ext_vector_type(8)))  float  v8f;

#define Hd    512
#define Bb    32
#define TBLK  16384          // bf16 elements per time block (B*H)
#define NWG   16             // persistent workgroups in the recurrence
#define RECT  512            // threads per recurrence workgroup (16 waves)
#define WSLICE (32 * 8 * 32 * 16)   // bf16 per WG weight slice (131072 = 256KB)

// Swizzled offset of element (batch b, hidden hh) inside one time block.
// Layout: [mt(2)][kt(16)][lane(32)][j(16)], matching the ISA 16-bit A-operand
// layout: lane = half*16 + (b%16); k_local = (j&7) + ((j>>3)<<4) + (half<<3).
__device__ __forceinline__ int swzb(int b, int hh) {
  int kl   = hh & 31;
  int j    = (((kl >> 4) & 1) << 3) | (kl & 7);
  int lane = (((kl >> 3) & 1) << 4) | (b & 15);
  return ((b >> 4) << 13) | ((hh >> 5) << 9) | (lane << 4) | j;
}

__device__ __forceinline__ float sigf(float x) { return 1.0f / (1.0f + __expf(-x)); }

// fp32 (B,T,H) row-major -> swizzled bf16 time blocks
__global__ void pack_x(const float* __restrict__ src, __bf16* __restrict__ dst, int T) {
  int idx = blockIdx.x * blockDim.x + threadIdx.x;
  int total = Bb * T * Hd;
  if (idx >= total) return;
  int perb = T * Hd;
  int b    = idx / perb;
  int rem  = idx - b * perb;
  int t    = rem >> 9;
  int hh   = rem & 511;
  dst[(size_t)t * TBLK + swzb(b, hh)] = (__bf16)src[idx];
}

// Pack [Wih;Whh]^T (K=1024, N=2048) into per-WG B-fragment order:
// dst layout [wg(16)][kt(32)][nt(8)][lane(32)][j(16)], 2M bf16 per layer.
// Slice column c = gate*32 + usub*16 + r -> global n = gate*512 + wg*32 + usub*16 + r.
__global__ void pack_W(const float* __restrict__ Wih, const float* __restrict__ Whh,
                       __bf16* __restrict__ dst) {
  int idx  = blockIdx.x * blockDim.x + threadIdx.x;   // exact grid: 2097152
  int j    = idx & 15;
  int lane = (idx >> 4) & 31;
  int nt   = (idx >> 9) & 7;
  int kt   = (idx >> 12) & 31;
  int wg   = idx >> 17;
  int r = lane & 15, half = lane >> 4;
  int gate = nt >> 1, usub = nt & 1;
  int n = gate * Hd + wg * 32 + usub * 16 + r;
  int k = kt * 32 + (j & 7) + ((j >> 3) << 4) + (half << 3);
  float v = (k < Hd) ? Wih[(size_t)n * Hd + k] : Whh[(size_t)n * Hd + (k - Hd)];
  dst[idx] = (__bf16)v;
}

// Pack upsample weight (H x N row-major) into B-fragment order [ntile][kt(16)][lane][j]
__global__ void pack_upW(const float* __restrict__ W, __bf16* __restrict__ dst, int N) {
  int idx  = blockIdx.x * blockDim.x + threadIdx.x;   // exact grid: 512*N
  int j    = idx & 15;
  int lane = (idx >> 4) & 31;
  int kt   = (idx >> 9) & 15;
  int nti  = idx >> 13;
  int r = lane & 15, half = lane >> 4;
  int k = kt * 32 + (j & 7) + ((j >> 3) << 4) + (half << 3);
  int n = nti * 16 + r;
  dst[idx] = (__bf16)W[(size_t)k * N + n];
}

__global__ void vec_add(const float* __restrict__ a, const float* __restrict__ b,
                        float* __restrict__ o, int n) {
  int i = blockIdx.x * blockDim.x + threadIdx.x;
  if (i < n) o[i] = a[i] + b[i];
}

// ---------------------------------------------------------------------------
// Persistent LSTM recurrence. Grid = NWG blocks of RECT threads (16 waves).
// Wave (mt, nt) computes the 16x16 tile (rows mt*16.., slice cols nt*16..) of
// gates(32 x 128) = [x_t ; h_{t-1}](32x1024) @ Wslice(1024x128), B from LDS.
// Each thread owns 2 (batch, unit) pairs: cell state lives in registers.
// ---------------------------------------------------------------------------
__global__ __launch_bounds__(RECT) void lstm_rec(
    const __bf16* __restrict__ xin,   // swizzled, T blocks (layer input)
    const __bf16* __restrict__ h0s,   // swizzled, 1 block  (initial h)
    const float*  __restrict__ c0,    // (B,H) fp32 row-major (initial c)
    const __bf16* __restrict__ Wp,    // packed weights (this layer)
    const float*  __restrict__ bias,  // (4,512) fp32, bih+bhh, gate-major
    __bf16*       __restrict__ hseq,  // out: swizzled, T blocks
    float*        __restrict__ outf,  // optional fp32 (B,T,H) row-major
    int T,
    unsigned*     __restrict__ bar)   // {count, generation}
{
  __shared__ __bf16 Wlds[WSLICE];               // 256 KB weight slice
  __shared__ float  gateLds[Bb * 128];          //  16 KB gate tile

  const int tid  = threadIdx.x, wg = blockIdx.x;
  const int lane = tid & 31,  wave = tid >> 5;
  const int mt   = wave >> 3, nt   = wave & 7;

  // One-time stage of this WG's weight slice into LDS (straight copy).
  {
    const uint4* src = (const uint4*)(Wp + (size_t)wg * WSLICE);
    uint4* dst4 = (uint4*)Wlds;
#pragma unroll
    for (int i = 0; i < 32; ++i)
      dst4[tid + i * RECT] = src[tid + i * RECT];
  }

  // Combine-phase ownership: 32 rows x 32 units / 512 threads = 2 pairs each.
  const int row0 = tid >> 4;
  const int u0   = (tid & 15) * 2;
  const int ug   = wg * 32 + u0;
  const float bi0 = bias[0 * Hd + ug], bi1 = bias[0 * Hd + ug + 1];
  const float bf0 = bias[1 * Hd + ug], bf1 = bias[1 * Hd + ug + 1];
  const float bg0 = bias[2 * Hd + ug], bg1 = bias[2 * Hd + ug + 1];
  const float bo0 = bias[3 * Hd + ug], bo1 = bias[3 * Hd + ug + 1];
  float ca = c0[(size_t)row0 * Hd + ug];
  float cb = c0[(size_t)row0 * Hd + ug + 1];
  const int so0 = swzb(row0, ug), so1 = swzb(row0, ug + 1);

  __syncthreads();

  for (int t = 0; t < T; ++t) {
    const __bf16* hprev = (t == 0) ? h0s : (hseq + (size_t)(t - 1) * TBLK);
    const __bf16* xb = xin   + (size_t)t * TBLK + mt * 8192 + lane * 16;
    const __bf16* hb = hprev + mt * 8192 + lane * 16;

    // Prefetch next timestep's x fragment region (global_prefetch_b8).
    __builtin_prefetch(xb + TBLK, 0, 1);

    v8f acc = {};
#pragma unroll
    for (int kt = 0; kt < 32; ++kt) {
      const __bf16* ap = (kt < 16) ? (xb + kt * 512) : (hb + (kt - 16) * 512);
      v16bf a = *(const v16bf*)ap;
      v16bf b = *(const v16bf*)&Wlds[((kt * 8 + nt) * 32 + lane) * 16];
      acc = __builtin_amdgcn_wmma_f32_16x16x32_bf16(false, a, false, b,
                                                    (short)0, acc, false, false);
    }
    // D fragment -> LDS gate tile: lane holds col n=lane&15, rows v + 8*(lane>>4)
    {
      int n  = lane & 15;
      int m0 = mt * 16 + ((lane >> 4) << 3);
#pragma unroll
      for (int v = 0; v < 8; ++v)
        gateLds[(m0 + v) * 128 + nt * 16 + n] = acc[v];
    }
    __syncthreads();
    // fp32 cell update for this thread's 2 (row, unit) pairs
    {
      float i0 = gateLds[row0 * 128 +      u0    ] + bi0;
      float f0 = gateLds[row0 * 128 + 32 + u0    ] + bf0;
      float g0 = gateLds[row0 * 128 + 64 + u0    ] + bg0;
      float o0 = gateLds[row0 * 128 + 96 + u0    ] + bo0;
      ca = sigf(f0) * ca + sigf(i0) * tanhf(g0);
      float hva = sigf(o0) * tanhf(ca);

      float i1 = gateLds[row0 * 128 +      u0 + 1] + bi1;
      float f1 = gateLds[row0 * 128 + 32 + u0 + 1] + bf1;
      float g1 = gateLds[row0 * 128 + 64 + u0 + 1] + bg1;
      float o1 = gateLds[row0 * 128 + 96 + u0 + 1] + bo1;
      cb = sigf(f1) * cb + sigf(i1) * tanhf(g1);
      float hvb = sigf(o1) * tanhf(cb);

      __bf16* hd = hseq + (size_t)t * TBLK;
      hd[so0] = (__bf16)hva;
      hd[so1] = (__bf16)hvb;
      if (outf) {
        float* od = outf + ((size_t)row0 * T + t) * Hd;
        od[ug]     = hva;
        od[ug + 1] = hvb;
      }
    }
    __threadfence();   // release our h stores (agent scope)
    __syncthreads();
    // Agent-scope grid barrier over NWG blocks (sense via generation counter).
    if (tid == 0) {
      unsigned my  = __hip_atomic_load(&bar[1], __ATOMIC_ACQUIRE, __HIP_MEMORY_SCOPE_AGENT);
      unsigned arr = __hip_atomic_fetch_add(&bar[0], 1u, __ATOMIC_ACQ_REL,
                                            __HIP_MEMORY_SCOPE_AGENT) + 1u;
      if (arr == NWG) {
        __hip_atomic_store(&bar[0], 0u, __ATOMIC_RELAXED, __HIP_MEMORY_SCOPE_AGENT);
        __hip_atomic_fetch_add(&bar[1], 1u, __ATOMIC_RELEASE, __HIP_MEMORY_SCOPE_AGENT);
      } else {
        while (__hip_atomic_load(&bar[1], __ATOMIC_ACQUIRE, __HIP_MEMORY_SCOPE_AGENT) == my)
          __builtin_amdgcn_s_sleep(2);
      }
    }
    __syncthreads();
    // Acquire: invalidate WGP-local caches so h_{t-1} loads see remote stores.
    __builtin_amdgcn_fence(__ATOMIC_ACQUIRE, "agent");
  }
}

// ---------------------------------------------------------------------------
// Upsampler: xin_next = x_next + reshape(h @ upW + upb).  WMMA GEMM with
// fused epilogue; grid (T, N/64), 8 waves per block (2 mt x 4 nt tiles).
// ---------------------------------------------------------------------------
__global__ __launch_bounds__(256) void upsample_wmma(
    const __bf16* __restrict__ hsw,   // swizzled h sequence, T blocks
    const __bf16* __restrict__ Wp,    // packed upW [N/16][16][32][16]
    const float*  __restrict__ bvec,  // (N)
    const float*  __restrict__ xnext, // fp32 (B, T*ups, H) row-major
    __bf16*       __restrict__ xout,  // swizzled, T*ups blocks
    int T, int ups)
{
  const int t   = blockIdx.x, nsl = blockIdx.y;
  const int tid = threadIdx.x, lane = tid & 31, wave = tid >> 5;
  const int mt  = wave >> 2, nt = wave & 3;
  const int ntile = nsl * 4 + nt;

  const __bf16* ab = hsw + (size_t)t * TBLK + mt * 8192 + lane * 16;
  const __bf16* bb = Wp  + (size_t)ntile * 8192 + (size_t)lane * 16;

  v8f acc = {};
#pragma unroll
  for (int kt = 0; kt < 16; ++kt) {
    v16bf a = *(const v16bf*)(ab + kt * 512);
    v16bf b = *(const v16bf*)(bb + (size_t)kt * 512);
    acc = __builtin_amdgcn_wmma_f32_16x16x32_bf16(false, a, false, b,
                                                  (short)0, acc, false, false);
  }
  const int n    = lane & 15;
  const int ncol = ntile * 16 + n;
  const int t2   = t * ups + (ncol >> 9);
  const int hh   = ncol & 511;
  const int T2   = T * ups;
  const float bv = bvec[ncol];
#pragma unroll
  for (int v = 0; v < 8; ++v) {
    int bat = mt * 16 + ((lane >> 4) << 3) + v;
    float val = acc[v] + bv + xnext[((size_t)bat * T2 + t2) * Hd + hh];
    xout[(size_t)t2 * TBLK + swzb(bat, hh)] = (__bf16)val;
  }
}

// ---------------------------------------------------------------------------
extern "C" void kernel_launch(void* const* d_in, const int* in_sizes, int n_in,
                              void* d_out, int out_size, void* d_ws, size_t ws_size,
                              hipStream_t stream) {
  (void)in_sizes; (void)n_in; (void)out_size; (void)ws_size;
  const int FR[3]   = {128, 256, 2048};
  const int UPS_[2] = {2, 8};

  char* wsb = (char*)d_ws;
  size_t off = 0;
  auto take = [&](size_t bytes) -> void* {
    void* p = wsb + off;
    off = (off + bytes + 255) & ~(size_t)255;
    return p;
  };
  unsigned* bar = (unsigned*)take(8);
  __bf16* xin  = (__bf16*)take((size_t)2048 * TBLK * 2);   // 64 MB (shared across tiers)
  __bf16* hsA  = (__bf16*)take((size_t)2048 * TBLK * 2);   // 64 MB
  __bf16* hsB  = (__bf16*)take((size_t)2048 * TBLK * 2);   // 64 MB
  __bf16* Wp[6];
  for (int i = 0; i < 6; ++i) Wp[i] = (__bf16*)take((size_t)2 * 1024 * 1024 * 2);
  __bf16* up0p = (__bf16*)take((size_t)512 * 1024 * 2);
  __bf16* up1p = (__bf16*)take((size_t)512 * 4096 * 2);
  float* biasb[6];
  for (int i = 0; i < 6; ++i) biasb[i] = (float*)take(2048 * 4);
  __bf16* h0b[6];
  for (int i = 0; i < 6; ++i) h0b[i] = (__bf16*)take((size_t)TBLK * 2);

  hipMemsetAsync(bar, 0, 8, stream);

  // ---- weight / state packing (parallel, cheap) ----
  for (int tier = 0; tier < 3; ++tier) {
    const float* Wih = (const float*)d_in[3 + tier * 6 + 0];
    const float* Whh = (const float*)d_in[3 + tier * 6 + 1];
    const float* bih = (const float*)d_in[3 + tier * 6 + 2];
    const float* bhh = (const float*)d_in[3 + tier * 6 + 3];
    const float* h0i = (const float*)d_in[3 + tier * 6 + 4];
    for (int l = 0; l < 2; ++l) {
      int id = tier * 2 + l;
      pack_W<<<2097152 / 256, 256, 0, stream>>>(Wih + (size_t)l * 2048 * 512,
                                                Whh + (size_t)l * 2048 * 512, Wp[id]);
      vec_add<<<8, 256, 0, stream>>>(bih + l * 2048, bhh + l * 2048, biasb[id], 2048);
      pack_x<<<64, 256, 0, stream>>>(h0i + (size_t)l * Bb * Hd, h0b[id], 1);
    }
  }
  pack_upW<<<512 * 1024 / 256, 256, 0, stream>>>((const float*)d_in[21], up0p, 1024);
  pack_upW<<<512 * 4096 / 256, 256, 0, stream>>>((const float*)d_in[23], up1p, 4096);
  pack_x<<<(Bb * 128 * Hd) / 256, 256, 0, stream>>>((const float*)d_in[0], xin, 128);

  // ---- three tiers: 2-layer LSTM + (upsample into next tier input) ----
  for (int tier = 0; tier < 3; ++tier) {
    const float* c0i = (const float*)d_in[3 + tier * 6 + 5];
    int T = FR[tier];
    lstm_rec<<<NWG, RECT, 0, stream>>>(xin, h0b[tier * 2 + 0], c0i,
                                       Wp[tier * 2 + 0], biasb[tier * 2 + 0],
                                       hsA, nullptr, T, bar);
    lstm_rec<<<NWG, RECT, 0, stream>>>(hsA, h0b[tier * 2 + 1], c0i + (size_t)Bb * Hd,
                                       Wp[tier * 2 + 1], biasb[tier * 2 + 1],
                                       hsB, (tier == 2) ? (float*)d_out : nullptr, T, bar);
    if (tier < 2) {
      int N = UPS_[tier] * Hd;
      dim3 g(T, N / 64);
      upsample_wmma<<<g, 256, 0, stream>>>(hsB, (tier == 0) ? up0p : up1p,
                                           (const float*)d_in[22 + tier * 2],
                                           (const float*)d_in[tier + 1],
                                           xin, T, UPS_[tier]);
    }
  }
}
